// Coattention_82798379532305
// MI455X (gfx1250) — matrix operand order repaired and checked by reading, not compiled, not run on hardware
//
#include <hip/hip_runtime.h>
#include <hip/hip_bf16.h>

// ---------------------------------------------------------------------------
// Coattention for MI455X (gfx1250, wave32, WMMA).
// All four 32x(1024^3) GEMMs run on v_wmma_f32_16x16x32_bf16 in NT form
// (both operands staged row-major, 16B chunks). GEMM1 uses bf16x3 split
// accumulation for ~fp32 logit precision. Tile staging uses the CDNA5
// async global->LDS path (ASYNCcnt) when the builtin is available.
// ---------------------------------------------------------------------------

typedef __attribute__((ext_vector_type(16))) __bf16 v16bf;
typedef __attribute__((ext_vector_type(8)))  __bf16 v8bf;
typedef __attribute__((ext_vector_type(4)))  __bf16 v4bf;
typedef __attribute__((ext_vector_type(8)))  float  v8f;
typedef __attribute__((ext_vector_type(4)))  int    v4i;

#define DEVI __device__ __forceinline__
#define AS1 __attribute__((address_space(1)))
#define AS3 __attribute__((address_space(3)))

#if defined(__has_builtin)
#  if __has_builtin(__builtin_amdgcn_global_load_async_to_lds_b128)
#    define HAVE_ASYNC_LDS 1
#  else
#    define HAVE_ASYNC_LDS 0
#  endif
#else
#  define HAVE_ASYNC_LDS 0
#endif

namespace {
constexpr int    kB     = 32;
constexpr int    kL     = 1024;                 // Lq == Ld == H
constexpr size_t kMat   = (size_t)kL * kL;      // per-batch matrix elements
constexpr size_t kElems = (size_t)kB * kMat;    // 33,554,432
constexpr int    BK     = 32;                   // K tile (matches WMMA K)
constexpr int    LDK    = BK + 8;               // padded LDS row (80B, 16B-aligned)
constexpr float  NEG_BIG = -1e30f;
}

// ---------------- CDNA5 helpers -------------------------------------------

DEVI void async_copy16(__bf16* lds, const __bf16* g) {
#if HAVE_ASYNC_LDS
  v4i* gg = (v4i*)const_cast<__bf16*>(g);   // generic ptr, int4 pointee
  v4i* ll = (v4i*)lds;
  __builtin_amdgcn_global_load_async_to_lds_b128((AS1 v4i*)gg, (AS3 v4i*)ll, 0, 0);
#else
  *(v8bf*)lds = *(const v8bf*)g;
#endif
}

DEVI void wait_async() {
#if HAVE_ASYNC_LDS
#  if __has_builtin(__builtin_amdgcn_s_wait_asynccnt)
  __builtin_amdgcn_s_wait_asynccnt(0);
#  else
  asm volatile("s_wait_asynccnt 0x0" ::: "memory");
#  endif
#endif
}

DEVI v8f wmma_bf16(v16bf a, v16bf b, v8f c) {
  // 8 args: (neg_a, A, neg_b, B, c_mod, C, reuse_a, reuse_b)
  return __builtin_amdgcn_wmma_f32_16x16x32_bf16(false, a, false, b,
                                                 (short)0, c, false, false);
}

// A fragment (16x32 bf16): lane holds row M=lane%16; K runs
// [(lane>>4)*8, +8) and [(lane>>4)*8+16, +8)  -> two 16B LDS reads.
DEVI v16bf load_frag_a(const __bf16* t, int row, int lane) {
  const __bf16* p = t + (row + (lane & 15)) * LDK + ((lane >> 4) << 3);
  v8bf lo = *(const v8bf*)p;
  v8bf hi = *(const v8bf*)(p + 16);
  return __builtin_shufflevector(lo, hi, 0,1,2,3,4,5,6,7,8,9,10,11,12,13,14,15);
}

// B fragment (32x16 bf16): lane holds col N=lane%16; K runs
// [(lane>>4)*8, +8) and [(lane>>4)*8+16, +8) -> two 16B LDS reads. (tile stored [N][K])
DEVI v16bf load_frag_b(const __bf16* t, int col, int lane) {
  const __bf16* p = t + (col + (lane & 15)) * LDK + ((lane >> 4) << 3);
  v8bf lo = *(const v8bf*)p;
  v8bf hi = *(const v8bf*)(p + 16);
  return __builtin_shufflevector(lo, hi, 0,1,2,3,4,5,6,7,8,9,10,11,12,13,14,15);
}

// ---------------- Kernel 1: fp32 -> (hi, lo) bf16 split -------------------

__global__ __launch_bounds__(256)
void split_bf16(const float* __restrict__ x, __bf16* __restrict__ hi,
                __bf16* __restrict__ lo) {
  size_t i4 = ((size_t)blockIdx.x * blockDim.x + threadIdx.x) * 4;
  float4 v = *(const float4*)(x + i4);
  v4bf h, l;
  float vv[4] = {v.x, v.y, v.z, v.w};
#pragma unroll
  for (int j = 0; j < 4; ++j) {
    __bf16 hb = (__bf16)vv[j];
    h[j] = hb;
    l[j] = (__bf16)(vv[j] - (float)hb);
  }
  *(v4bf*)(hi + i4) = h;
  *(v4bf*)(lo + i4) = l;
}

// ---------------- Kernel 2: a = q . d^T  (bf16x3 split, NT) ---------------
// Single LDS buffer set (40KB) + register prefetch of next-k global tiles.

__global__ __launch_bounds__(256)
void gemm_nt_split(const __bf16* __restrict__ Ahi, const __bf16* __restrict__ Alo,
                   const __bf16* __restrict__ Bhi, const __bf16* __restrict__ Blo,
                   float* __restrict__ C) {
  __shared__ __align__(16) __bf16 Ash[128 * LDK];
  __shared__ __align__(16) __bf16 Asl[128 * LDK];
  __shared__ __align__(16) __bf16 Bsh[128 * LDK];
  __shared__ __align__(16) __bf16 Bsl[128 * LDK];

  const int tid  = threadIdx.x;
  const int lane = tid & 31;
  const int wave = tid >> 5;
  const int wm = wave >> 1, wn = wave & 1;     // 4x2 wave grid
  const int b  = blockIdx.z;
  const int m0 = blockIdx.y * 128, n0 = blockIdx.x * 128;

  const __bf16* Ah = Ahi + (size_t)b * kMat + (size_t)m0 * kL;
  const __bf16* Al = Alo + (size_t)b * kMat + (size_t)m0 * kL;
  const __bf16* Bh = Bhi + (size_t)b * kMat + (size_t)n0 * kL;  // [N][K] row-major
  const __bf16* Bl = Blo + (size_t)b * kMat + (size_t)n0 * kL;

  // two 16B chunks per thread per tile
  const int ca = tid, cb = tid + 256;
  const int ra = ca >> 2, ka = (ca & 3) << 3;
  const int rb = cb >> 2, kb = (cb & 3) << 3;

  v8bf pre[8];
  auto gload = [&](int k0) {
    pre[0] = *(const v8bf*)(Ah + (size_t)ra * kL + k0 + ka);
    pre[1] = *(const v8bf*)(Ah + (size_t)rb * kL + k0 + kb);
    pre[2] = *(const v8bf*)(Al + (size_t)ra * kL + k0 + ka);
    pre[3] = *(const v8bf*)(Al + (size_t)rb * kL + k0 + kb);
    pre[4] = *(const v8bf*)(Bh + (size_t)ra * kL + k0 + ka);
    pre[5] = *(const v8bf*)(Bh + (size_t)rb * kL + k0 + kb);
    pre[6] = *(const v8bf*)(Bl + (size_t)ra * kL + k0 + ka);
    pre[7] = *(const v8bf*)(Bl + (size_t)rb * kL + k0 + kb);
  };
  auto lstore = [&]() {
    *(v8bf*)(Ash + ra * LDK + ka) = pre[0];
    *(v8bf*)(Ash + rb * LDK + kb) = pre[1];
    *(v8bf*)(Asl + ra * LDK + ka) = pre[2];
    *(v8bf*)(Asl + rb * LDK + kb) = pre[3];
    *(v8bf*)(Bsh + ra * LDK + ka) = pre[4];
    *(v8bf*)(Bsh + rb * LDK + kb) = pre[5];
    *(v8bf*)(Bsl + ra * LDK + ka) = pre[6];
    *(v8bf*)(Bsl + rb * LDK + kb) = pre[7];
  };

  v8f acc[2][4];
  const v8f vzero = {0.f,0.f,0.f,0.f,0.f,0.f,0.f,0.f};
#pragma unroll
  for (int i = 0; i < 2; ++i)
#pragma unroll
    for (int j = 0; j < 4; ++j) acc[i][j] = vzero;

  gload(0);
  for (int k0 = 0; k0 < kL; k0 += BK) {
    lstore();
    __syncthreads();
    int kn = (k0 + BK < kL) ? k0 + BK : 0;
    gload(kn);  // overlap next-k global loads with WMMA below

    v16bf ah[2], al[2], bh[4], bl[4];
#pragma unroll
    for (int i = 0; i < 2; ++i) {
      ah[i] = load_frag_a(Ash, wm * 32 + i * 16, lane);
      al[i] = load_frag_a(Asl, wm * 32 + i * 16, lane);
    }
#pragma unroll
    for (int j = 0; j < 4; ++j) {
      bh[j] = load_frag_b(Bsh, wn * 64 + j * 16, lane);
      bl[j] = load_frag_b(Bsl, wn * 64 + j * 16, lane);
    }
#pragma unroll
    for (int i = 0; i < 2; ++i)
#pragma unroll
      for (int j = 0; j < 4; ++j) {
        acc[i][j] = wmma_bf16(ah[i], bh[j], acc[i][j]);
        acc[i][j] = wmma_bf16(ah[i], bl[j], acc[i][j]);
        acc[i][j] = wmma_bf16(al[i], bh[j], acc[i][j]);
      }
    __syncthreads();
  }

  float* Cb = C + (size_t)b * kMat;
  const int nloc = lane & 15, mhi = (lane >> 4) << 3;
#pragma unroll
  for (int i = 0; i < 2; ++i)
#pragma unroll
    for (int j = 0; j < 4; ++j) {
      int mBase = m0 + wm * 32 + i * 16 + mhi;
      int n     = n0 + wn * 64 + j * 16 + nloc;
#pragma unroll
      for (int r = 0; r < 8; ++r)
        Cb[(size_t)(mBase + r) * kL + n] = acc[i][j][r];
    }
}

// ---------------- Kernel 3/4: softmax statistics --------------------------

__global__ __launch_bounds__(256)
void col_stats(const float* __restrict__ a, const int* __restrict__ q_len,
               float* __restrict__ cmax, float* __restrict__ csum) {
  int b = blockIdx.y;
  int d = blockIdx.x * 256 + threadIdx.x;
  int ql = q_len[b];
  const float* ab = a + (size_t)b * kMat;
  float m = NEG_BIG, s = 0.f;
  for (int q = 0; q < ql; ++q) {
    float x = ab[(size_t)q * kL + d];
    if (x > m) { s = s * __expf(m - x) + 1.f; m = x; }
    else       { s += __expf(x - m); }
  }
  cmax[b * kL + d] = m;
  csum[b * kL + d] = s;
}

__global__ __launch_bounds__(256)
void row_stats(const float* __restrict__ a, const int* __restrict__ d_len,
               float* __restrict__ rmax, float* __restrict__ rsum) {
  int b = blockIdx.y;
  int q = blockIdx.x * 8 + (threadIdx.x >> 5);
  int lane = threadIdx.x & 31;
  int dl = d_len[b];
  const float* row = a + (size_t)b * kMat + (size_t)q * kL;
  float m = NEG_BIG, s = 0.f;
  for (int d = lane; d < dl; d += 32) {
    float x = row[d];
    if (x > m) { s = s * __expf(m - x) + 1.f; m = x; }
    else       { s += __expf(x - m); }
  }
#pragma unroll
  for (int off = 16; off > 0; off >>= 1) {
    float om = __shfl_xor(m, off, 32);
    float os = __shfl_xor(s, off, 32);
    float M  = fmaxf(m, om);
    s = s * __expf(m - M) + os * __expf(om - M);
    m = M;
  }
  if (lane == 0) { rmax[b * kL + q] = m; rsum[b * kL + q] = s; }
}

// ---------------- Kernel 5: ar = row-softmax(a) in bf16 -------------------

__global__ __launch_bounds__(256)
void mat_row_softmax(const float* __restrict__ a, const float* __restrict__ rmax,
                     const float* __restrict__ rsum, const int* __restrict__ d_len,
                     __bf16* __restrict__ ar) {
  size_t base = ((size_t)blockIdx.x * 256 + threadIdx.x) * 4;
  int b  = (int)(base >> 20);
  int q  = (int)((base >> 10) & (kL - 1));
  int d0 = (int)(base & (kL - 1));
  int dl = d_len[b];
  float m   = rmax[b * kL + q];
  float inv = 1.f / rsum[b * kL + q];
  float4 x = *(const float4*)(a + base);
  float vv[4] = {x.x, x.y, x.z, x.w};
  v4bf o;
#pragma unroll
  for (int j = 0; j < 4; ++j)
    o[j] = (__bf16)((d0 + j < dl) ? __expf(vv[j] - m) * inv : 0.f);
  *(v4bf*)(ar + base) = o;
}

// ---------------- Kernel 6: aqT = col-softmax(a)^T in bf16 ----------------

__global__ __launch_bounds__(256)
void mat_col_softmax_T(const float* __restrict__ a, const float* __restrict__ cmax,
                       const float* __restrict__ csum, const int* __restrict__ q_len,
                       __bf16* __restrict__ aqT) {
  __shared__ float tile[32][33];
  int b  = blockIdx.z;
  int q0 = blockIdx.x * 32, d0 = blockIdx.y * 32;
  int tx = threadIdx.x & 31, ty = threadIdx.x >> 5;   // 32 x 8
  const float* ab = a + (size_t)b * kMat;
#pragma unroll
  for (int k = 0; k < 4; ++k)
    tile[ty + 8 * k][tx] = ab[(size_t)(q0 + ty + 8 * k) * kL + (d0 + tx)];
  __syncthreads();
  int ql = q_len[b];
  __bf16* ob = aqT + (size_t)b * kMat;
#pragma unroll
  for (int k = 0; k < 4; ++k) {
    int d = d0 + ty + 8 * k;
    int q = q0 + tx;
    float m = cmax[b * kL + d], inv = 1.f / csum[b * kL + d];
    float v = (q < ql) ? __expf(tile[tx][ty + 8 * k] - m) * inv : 0.f;
    ob[(size_t)d * kL + q] = (__bf16)v;
  }
}

// ---------------- Kernel 7: bf16 LxL transpose (per batch) ----------------

__global__ __launch_bounds__(256)
void transpose_bf16(const __bf16* __restrict__ src, __bf16* __restrict__ dst) {
  __shared__ __bf16 t[32][33];
  int b  = blockIdx.z;
  int r0 = blockIdx.x * 32, c0 = blockIdx.y * 32;
  int tx = threadIdx.x & 31, ty = threadIdx.x >> 5;
  const __bf16* sb = src + (size_t)b * kMat;
  __bf16* db = dst + (size_t)b * kMat;
#pragma unroll
  for (int k = 0; k < 4; ++k)
    t[ty + 8 * k][tx] = sb[(size_t)(r0 + ty + 8 * k) * kL + (c0 + tx)];
  __syncthreads();
#pragma unroll
  for (int k = 0; k < 4; ++k)
    db[(size_t)(c0 + ty + 8 * k) * kL + (r0 + tx)] = t[tx][ty + 8 * k];
}

// ---------------- Kernel 8: C = A . B^T (bf16 NT) -------------------------
// Double-buffered LDS, async global->LDS staging, one barrier per k-step.
// Optional transposed bf16 copy of C (contiguous v8bf stores per lane).

__global__ __launch_bounds__(256)
void gemm_nt_bf16(const __bf16* __restrict__ A, const __bf16* __restrict__ B,
                  float* __restrict__ C, __bf16* __restrict__ CbfT) {
  __shared__ __align__(16) __bf16 As[2][128 * LDK];
  __shared__ __align__(16) __bf16 Bs[2][128 * LDK];

  const int tid  = threadIdx.x;
  const int lane = tid & 31;
  const int wave = tid >> 5;
  const int wm = wave >> 1, wn = wave & 1;
  const int b  = blockIdx.z;
  const int m0 = blockIdx.y * 128, n0 = blockIdx.x * 128;

  const __bf16* Ab = A + (size_t)b * kMat + (size_t)m0 * kL;
  const __bf16* Bb = B + (size_t)b * kMat + (size_t)n0 * kL;  // [N][K] row-major

  const int ca = tid, cb = tid + 256;
  const int ra = ca >> 2, ka = (ca & 3) << 3;
  const int rb = cb >> 2, kb = (cb & 3) << 3;

  auto stage = [&](int buf, int k0) {
    async_copy16(&As[buf][ra * LDK + ka], Ab + (size_t)ra * kL + k0 + ka);
    async_copy16(&As[buf][rb * LDK + kb], Ab + (size_t)rb * kL + k0 + kb);
    async_copy16(&Bs[buf][ra * LDK + ka], Bb + (size_t)ra * kL + k0 + ka);
    async_copy16(&Bs[buf][rb * LDK + kb], Bb + (size_t)rb * kL + k0 + kb);
  };

  v8f acc[2][4];
  const v8f vzero = {0.f,0.f,0.f,0.f,0.f,0.f,0.f,0.f};
#pragma unroll
  for (int i = 0; i < 2; ++i)
#pragma unroll
    for (int j = 0; j < 4; ++j) acc[i][j] = vzero;

  stage(0, 0);
  wait_async();
  __syncthreads();

  for (int k0 = 0, cur = 0; k0 < kL; k0 += BK, cur ^= 1) {
    if (k0 + BK < kL) stage(cur ^ 1, k0 + BK);  // fill other buffer while computing

    v16bf af[2], bfr[4];
#pragma unroll
    for (int i = 0; i < 2; ++i) af[i] = load_frag_a(As[cur], wm * 32 + i * 16, lane);
#pragma unroll
    for (int j = 0; j < 4; ++j) bfr[j] = load_frag_b(Bs[cur], wn * 64 + j * 16, lane);
#pragma unroll
    for (int i = 0; i < 2; ++i)
#pragma unroll
      for (int j = 0; j < 4; ++j)
        acc[i][j] = wmma_bf16(af[i], bfr[j], acc[i][j]);

    wait_async();
    __syncthreads();
  }

  float* Cb = C + (size_t)b * kMat;
  const int nloc = lane & 15, mhi = (lane >> 4) << 3;
#pragma unroll
  for (int i = 0; i < 2; ++i)
#pragma unroll
    for (int j = 0; j < 4; ++j) {
      int mBase = m0 + wm * 32 + i * 16 + mhi;
      int n     = n0 + wn * 64 + j * 16 + nloc;
#pragma unroll
      for (int r = 0; r < 8; ++r)
        Cb[(size_t)(mBase + r) * kL + n] = acc[i][j][r];
    }
  if (CbfT) {
    __bf16* Ct = CbfT + (size_t)b * kMat;
#pragma unroll
    for (int i = 0; i < 2; ++i)
#pragma unroll
      for (int j = 0; j < 4; ++j) {
        int mBase = m0 + wm * 32 + i * 16 + mhi;
        int n     = n0 + wn * 64 + j * 16 + nloc;
        v8bf t;
#pragma unroll
        for (int r = 0; r < 8; ++r) t[r] = (__bf16)acc[i][j][r];
        *(v8bf*)(Ct + (size_t)n * kL + mBase) = t;  // transposed: 16B contiguous
      }
  }
}

// ---------------- Host orchestration --------------------------------------

extern "C" void kernel_launch(void* const* d_in, const int* in_sizes, int n_in,
                              void* d_out, int out_size, void* d_ws, size_t ws_size,
                              hipStream_t stream) {
  (void)in_sizes; (void)n_in; (void)out_size; (void)ws_size;
  const float* q     = (const float*)d_in[0];
  const float* d     = (const float*)d_in[1];
  const int*   q_len = (const int*)d_in[2];
  const int*   d_len = (const int*)d_in[3];

  // Workspace (~384.5 MiB): ws0..ws3 = 4 x 64 MiB bf16, a = 128 MiB fp32
  // (L2-resident), stats = 4 x 128 KiB fp32. Reuse after GEMM1:
  //   qT -> ws1 (qlo), dT -> ws3 (dlo), ar -> ws0 (qhi), aqT -> ws2 (dhi),
  //   sq^T bf16 (transposed) -> a's space.
  __bf16* ws0 = (__bf16*)d_ws;
  __bf16* ws1 = ws0 + kElems;
  __bf16* ws2 = ws1 + kElems;
  __bf16* ws3 = ws2 + kElems;
  float*  a   = (float*)(ws3 + kElems);
  float*  cmax = a + kElems;
  float*  csum = cmax + (size_t)kB * kL;
  float*  rmax = csum + (size_t)kB * kL;
  float*  rsum = rmax + (size_t)kB * kL;

  __bf16* qhi = ws0, *qlo = ws1, *dhi = ws2, *dlo = ws3;
  __bf16* qT  = ws1, *dT  = ws3;          // overwrite lo parts after GEMM1
  __bf16* ar  = ws0, *aqT = ws2;          // overwrite hi parts after transposes
  __bf16* sqT = (__bf16*)a;               // overwrite a after softmax kernels

  float* out_cd = (float*)d_out;          // cd^T [B, Ld, H]
  float* out_sq = out_cd + kElems;        // sq^T [B, Lq, H]
  float* out_sd = out_sq + kElems;        // sd^T [B, Ld, H]

  const int cvtBlocks = (int)(kElems / 4 / 256);      // 32768
  split_bf16<<<cvtBlocks, 256, 0, stream>>>(q, qhi, qlo);
  split_bf16<<<cvtBlocks, 256, 0, stream>>>(d, dhi, dlo);

  dim3 gGemm(kL / 128, kL / 128, kB);                 // (8, 8, 32)
  gemm_nt_split<<<gGemm, 256, 0, stream>>>(qhi, qlo, dhi, dlo, a);

  dim3 gTr(kL / 32, kL / 32, kB);
  transpose_bf16<<<gTr, 256, 0, stream>>>(qhi, qT);   // qT overwrites qlo
  transpose_bf16<<<gTr, 256, 0, stream>>>(dhi, dT);   // dT overwrites dlo

  col_stats<<<dim3(kL / 256, kB), 256, 0, stream>>>(a, q_len, cmax, csum);
  row_stats<<<dim3(kL / 8,  kB), 256, 0, stream>>>(a, d_len, rmax, rsum);

  mat_row_softmax<<<cvtBlocks, 256, 0, stream>>>(a, rmax, rsum, d_len, ar);
  mat_col_softmax_T<<<gTr, 256, 0, stream>>>(a, cmax, csum, q_len, aqT);

  // sq^T = ar . dT^T       (+ transposed bf16 copy for the cd GEMM)
  gemm_nt_bf16<<<gGemm, 256, 0, stream>>>(ar, dT, out_sq, sqT);
  // sd^T = aqT . qT^T
  gemm_nt_bf16<<<gGemm, 256, 0, stream>>>(aqT, qT, out_sd, nullptr);
  // cd^T = aqT . (sq^T)^T
  gemm_nt_bf16<<<gGemm, 256, 0, stream>>>(aqT, sqT, out_cd, nullptr);
}
